// GraphConvolution_28037546508960
// MI455X (gfx1250) — matrix operand order repaired and checked by reading, not compiled, not run on hardware
//
#include <hip/hip_runtime.h>

#define N_NODES 8192
#define IN_DIM  64
#define OUT_DIM 64
#define SUPPORT 4
#define NUM_BASES 2

typedef __bf16 bh16;
typedef __attribute__((ext_vector_type(16))) __bf16 v16bf;
typedef __attribute__((ext_vector_type(8)))  __bf16 bh16x8;
typedef __attribute__((ext_vector_type(8)))  float  v8f;
typedef __attribute__((ext_vector_type(4)))  float  f32x4;

#define KSTAGE  64                         // K depth staged in LDS per buffer
#define NSTAGES ((N_NODES / 2) / KSTAGE)   // 64 stages per (s, K-half) split
#define ZPAD    72                         // 64 + 8 bf16 pad: 144B rows, conflict-free

// ---------------------------------------------------------------------------
// Prep 1: W[s][d][n] = comp[s,0]*kernel[0][d][n] + comp[s,1]*kernel[1][d][n]
// ---------------------------------------------------------------------------
__global__ void compute_W_kernel(const float* __restrict__ kern,
                                 const float* __restrict__ comp,
                                 float* __restrict__ W) {
    int idx = blockIdx.x * 256 + threadIdx.x;          // SUPPORT*64*64 = 16384
    int s   = idx >> 12;
    int dn  = idx & 4095;
    W[idx] = comp[s * NUM_BASES + 0] * kern[dn] +
             comp[s * NUM_BASES + 1] * kern[IN_DIM * OUT_DIM + dn];
}

// ---------------------------------------------------------------------------
// Prep 2: Zt[s][n][k] = sum_d X[k][d] * W[s][d][n], stored bf16 transposed.
// ---------------------------------------------------------------------------
__global__ void compute_Zt_kernel(const float* __restrict__ X,
                                  const float* __restrict__ W,
                                  bh16* __restrict__ Zt) {
    int bid = blockIdx.x;            // SUPPORT * N_NODES blocks
    int s   = bid >> 13;
    int k   = bid & (N_NODES - 1);
    __shared__ float xrow[IN_DIM];
    xrow[threadIdx.x] = X[(size_t)k * IN_DIM + threadIdx.x];
    __syncthreads();
    int n = threadIdx.x;
    const float* Ws = W + (size_t)s * IN_DIM * OUT_DIM;
    float sum = 0.f;
#pragma unroll
    for (int d = 0; d < IN_DIM; ++d) sum += xrow[d] * Ws[d * OUT_DIM + n];
    Zt[((size_t)(s * OUT_DIM + n)) * N_NODES + k] = (bh16)sum;
}

// ---------------------------------------------------------------------------
// Main skinny GEMM: acc += A_s @ Z_s, bf16 WMMA 16x16x32, fp32 accum.
// A streamed from HBM (fp32 -> bf16 in regs); Z staged in LDS via
// global_load_async_to_lds_b128 with double buffering.
// ---------------------------------------------------------------------------
__device__ __forceinline__ v16bf load_bfrag(const bh16* __restrict__ p) {
    bh16x8 lo = *(const bh16x8*)(p + 0);
    bh16x8 hi = *(const bh16x8*)(p + 16);
    v16bf b;
#pragma unroll
    for (int i = 0; i < 8; ++i) { b[i] = lo[i]; b[8 + i] = hi[i]; }
    return b;
}

__global__ __launch_bounds__(256) void rgcn_spmm_kernel(
    const float* __restrict__ A, const bh16* __restrict__ Zt,
    float* __restrict__ acc) {
    __shared__ __align__(16) bh16 zbuf[2][OUT_DIM][ZPAD];

    const int t     = threadIdx.x;
    const int lane  = t & 31;
    const int wave  = t >> 5;
    const int m0    = blockIdx.x * 128 + wave * 16;
    const int s     = blockIdx.y >> 1;
    const int k0    = (blockIdx.y & 1) * (N_NODES / 2);

    const int l15    = lane & 15;
    const int half8  = (lane >> 4) << 3;   // A frag: K = h*8+[0..7], 16+h*8+[0..7]
    const int half16 = (lane >> 4) << 4;   // unused; B frag uses half8, same K order as A

    const float* __restrict__ Arow =
        A + (size_t)s * N_NODES * N_NODES + (size_t)(m0 + l15) * N_NODES;
    const bh16* __restrict__ Zs = Zt + (size_t)s * OUT_DIM * N_NODES;

    // Async-copy assignment: 512 x 16B pieces per stage, 2 per thread.
    const int cn   = t >> 3;        // rows cn and cn+32
    const int cseg = (t & 7) * 8;   // bf16 element offset within row
    const uint32_t zbase = (uint32_t)(uintptr_t)(&zbuf[0][0][0]);

    auto issue_stage = [&](int st, int buf) {
        const int kc = k0 + st * KSTAGE;
        const bh16* g0 = Zs + (size_t)cn * N_NODES + kc + cseg;
        const bh16* g1 = Zs + (size_t)(cn + 32) * N_NODES + kc + cseg;
        uint32_t l0 = zbase + (uint32_t)(((buf * OUT_DIM + cn) * ZPAD + cseg) * 2);
        uint32_t l1 = zbase + (uint32_t)(((buf * OUT_DIM + cn + 32) * ZPAD + cseg) * 2);
        asm volatile("global_load_async_to_lds_b128 %0, %1, off"
                     :: "v"(l0), "v"(g0) : "memory");
        asm volatile("global_load_async_to_lds_b128 %0, %1, off"
                     :: "v"(l1), "v"(g1) : "memory");
    };

    v8f c0 = {}, c1 = {}, c2 = {}, c3 = {};

    issue_stage(0, 0);
    for (int st = 0; st < NSTAGES; ++st) {
        const int buf = st & 1;
        if (st + 1 < NSTAGES) {
            issue_stage(st + 1, buf ^ 1);
            asm volatile("s_wait_asynccnt 0x2" ::: "memory");  // stage st complete
        } else {
            asm volatile("s_wait_asynccnt 0x0" ::: "memory");
        }
        __syncthreads();

#pragma unroll
        for (int step = 0; step < 2; ++step) {
            const int kc = k0 + st * KSTAGE + step * 32;

            // ---- A fragment: 16 fp32 -> bf16 in registers ----
            const float* ap = Arow + kc + half8;
            f32x4 f0 = *(const f32x4*)(ap + 0);
            f32x4 f1 = *(const f32x4*)(ap + 4);
            f32x4 f2 = *(const f32x4*)(ap + 16);
            f32x4 f3 = *(const f32x4*)(ap + 20);
            __builtin_prefetch((const void*)(ap + 256), 0, 3);  // ~1KB ahead

            v16bf a;
#pragma unroll
            for (int i = 0; i < 4; ++i) {
                a[i]      = (bh16)f0[i];
                a[4 + i]  = (bh16)f1[i];
                a[8 + i]  = (bh16)f2[i];
                a[12 + i] = (bh16)f3[i];
            }

            // ---- B fragments from LDS (ds_load_b128, conflict-free) ----
            const int kloc = step * 32 + half8;
            v16bf b0 = load_bfrag(&zbuf[buf][0 * 16 + l15][kloc]);
            v16bf b1 = load_bfrag(&zbuf[buf][1 * 16 + l15][kloc]);
            v16bf b2 = load_bfrag(&zbuf[buf][2 * 16 + l15][kloc]);
            v16bf b3 = load_bfrag(&zbuf[buf][3 * 16 + l15][kloc]);

            c0 = __builtin_amdgcn_wmma_f32_16x16x32_bf16(false, a, false, b0,
                                                         (short)0, c0, false, false);
            c1 = __builtin_amdgcn_wmma_f32_16x16x32_bf16(false, a, false, b1,
                                                         (short)0, c1, false, false);
            c2 = __builtin_amdgcn_wmma_f32_16x16x32_bf16(false, a, false, b2,
                                                         (short)0, c2, false, false);
            c3 = __builtin_amdgcn_wmma_f32_16x16x32_bf16(false, a, false, b3,
                                                         (short)0, c3, false, false);
        }
        __syncthreads();   // all reads of this buffer done before it is rewritten
    }

    // C/D layout: lanes 0-15 -> N=lane, VGPR r -> M=r; lanes 16-31 -> M=8+r.
    float* arow = acc + (size_t)(m0 + (lane >> 4) * 8) * OUT_DIM + l15;
#pragma unroll
    for (int r = 0; r < 8; ++r) {
        float* p = arow + (size_t)r * OUT_DIM;
        (void)__hip_atomic_fetch_add(p + 0,  c0[r], __ATOMIC_RELAXED, __HIP_MEMORY_SCOPE_AGENT);
        (void)__hip_atomic_fetch_add(p + 16, c1[r], __ATOMIC_RELAXED, __HIP_MEMORY_SCOPE_AGENT);
        (void)__hip_atomic_fetch_add(p + 32, c2[r], __ATOMIC_RELAXED, __HIP_MEMORY_SCOPE_AGENT);
        (void)__hip_atomic_fetch_add(p + 48, c3[r], __ATOMIC_RELAXED, __HIP_MEMORY_SCOPE_AGENT);
    }
}

// ---------------------------------------------------------------------------
// Epilogue: out = relu(acc + bias)
// ---------------------------------------------------------------------------
__global__ void bias_relu_kernel(const float* __restrict__ acc,
                                 const float* __restrict__ bias,
                                 float* __restrict__ out) {
    int idx = blockIdx.x * 256 + threadIdx.x;
    float v = acc[idx] + bias[idx & (OUT_DIM - 1)];
    out[idx] = v > 0.f ? v : 0.f;
}

extern "C" void kernel_launch(void* const* d_in, const int* in_sizes, int n_in,
                              void* d_out, int out_size, void* d_ws, size_t ws_size,
                              hipStream_t stream) {
    const float* features = (const float*)d_in[0];   // [8192,64]
    const float* A        = (const float*)d_in[1];   // [4,8192,8192]
    const float* kern     = (const float*)d_in[2];   // [128,64]
    const float* comp     = (const float*)d_in[3];   // [4,2]
    const float* bias     = (const float*)d_in[4];   // [64]
    float* out = (float*)d_out;                      // [8192,64]

    char*  ws  = (char*)d_ws;
    float* acc = (float*)ws;                                         // 2 MiB
    size_t off = (size_t)N_NODES * OUT_DIM * sizeof(float);
    float* W   = (float*)(ws + off);                                 // 64 KiB
    off += (size_t)SUPPORT * IN_DIM * OUT_DIM * sizeof(float);
    bh16*  Zt  = (bh16*)(ws + off);                                  // 4 MiB

    hipMemsetAsync(acc, 0, (size_t)N_NODES * OUT_DIM * sizeof(float), stream);

    compute_W_kernel<<<(SUPPORT * IN_DIM * OUT_DIM) / 256, 256, 0, stream>>>(kern, comp, W);
    compute_Zt_kernel<<<SUPPORT * N_NODES, IN_DIM, 0, stream>>>(features, W, Zt);
    rgcn_spmm_kernel<<<dim3(N_NODES / 128, 8), 256, 0, stream>>>(A, Zt, acc);
    bias_relu_kernel<<<(N_NODES * OUT_DIM) / 256, 256, 0, stream>>>(acc, bias, out);
}
